// AttentionPropagation_13443247636678
// MI455X (gfx1250) — matrix operand order repaired and checked by reading, not compiled, not run on hardware
//
#include <hip/hip_runtime.h>
#include <hip/hip_bf16.h>

// ---------------------------------------------------------------------------
// Types
// ---------------------------------------------------------------------------
typedef __bf16 bf16_t;
typedef __attribute__((ext_vector_type(16))) __bf16 v16bf;
typedef __attribute__((ext_vector_type(8)))  __bf16 v8bf;
typedef __attribute__((ext_vector_type(8)))  float  v8f;

#define BATCH 8
#define CCH   256
#define NPOS  4096
#define NHEAD 4

// ---------------------------------------------------------------------------
// WMMA helpers (CDNA5 wave32, v_wmma_f32_16x16x32_bf16)
//
// Fragment layouts (per CDNA5 ISA):
//   A (16x32, bf16): lane r (row M=r%16), half hh=lane>=16:
//       element e -> K = e + 8*(e>=8) + 8*hh   (two contiguous 8-element runs)
//   B (32x16, bf16): lane col = lane%16, half hh:
//       element e -> K = e + 8*(e>=8) + 8*hh   (two contiguous 8-element runs)
//   C/D (16x16 f32): VGPR j -> row M = j + 8*hh, col N = lane%16
// ---------------------------------------------------------------------------
__device__ __forceinline__ v8f wmma_bf16(v16bf a, v16bf b, v8f c) {
  return __builtin_amdgcn_wmma_f32_16x16x32_bf16(
      /*neg_a=*/false, a, /*neg_b=*/false, b,
      /*c_mod=*/(short)0, c, /*reuse_a=*/false, /*reuse_b=*/false);
}

__device__ __forceinline__ v16bf frag_2runs(const bf16_t* p0, const bf16_t* p1) {
  v8bf lo = *reinterpret_cast<const v8bf*>(p0);
  v8bf hi = *reinterpret_cast<const v8bf*>(p1);
  return __builtin_shufflevector(lo, hi, 0, 1, 2, 3, 4, 5, 6, 7,
                                 8, 9, 10, 11, 12, 13, 14, 15);
}
__device__ __forceinline__ v16bf frag_1run(const bf16_t* p) {
  return frag_2runs(p, p + 8);
}

// ---------------------------------------------------------------------------
// Elementwise f32 -> bf16 cast (for the small weight matrices)
// ---------------------------------------------------------------------------
__global__ void cast_f32_bf16_kernel(const float* __restrict__ src,
                                     bf16_t* __restrict__ dst, int n) {
  for (int i = blockIdx.x * blockDim.x + threadIdx.x; i < n;
       i += gridDim.x * blockDim.x)
    dst[i] = (bf16_t)src[i];
}

// ---------------------------------------------------------------------------
// [B,C,N] (TIN) -> [B,N,ncStride] bf16 transpose via LDS tile.
// AFFINE: y = relu(x*scale[c] + shift[c])  (fused BatchNorm+ReLU).
// ---------------------------------------------------------------------------
template <typename TIN, bool AFFINE>
__global__ __launch_bounds__(256)
void transpose_cn_nc_kernel(const TIN* __restrict__ in, bf16_t* __restrict__ out,
                            const float* __restrict__ scaleArr,
                            const float* __restrict__ shiftArr,
                            int C, int Npos, int ncStride) {
  __shared__ bf16_t tile[32][34];
  const int b   = blockIdx.z;
  const int c0  = blockIdx.y * 32;
  const int nt  = blockIdx.x * 32;
  const int col = threadIdx.x & 31;
  const int r   = threadIdx.x >> 5;  // 0..7
#pragma unroll
  for (int i = 0; i < 4; ++i) {
    const int c = c0 + r + i * 8;
    float v = (float)in[((size_t)b * C + c) * Npos + nt + col];
    if (AFFINE) v = fmaxf(v * scaleArr[c] + shiftArr[c], 0.f);
    tile[r + i * 8][col] = (bf16_t)v;
  }
  __syncthreads();
#pragma unroll
  for (int i = 0; i < 4; ++i) {
    const int nl = r + i * 8;
    out[((size_t)b * Npos + nt + nl) * ncStride + c0 + col] = tile[col][nl];
  }
}

// ---------------------------------------------------------------------------
// Generic bf16 WMMA GEMM:  Y[b] = W(Cout x Cin) * X[b](Cin x Npos) + bias
// X stored position-major: X[b][n][cin] (leading dim xld), bf16.
// One wave computes a 64(Cout) x 16(N) tile (4 accumulators, B-frag reuse x4).
// Store modes:
//   0: f32 [B,Cout,N] with residual add           (final output)
//   1: bf16 [B,Cout,N] channel-major              (V, h)
//   2: bf16 [B,N,ncStride] position-major         (wm -> concat buffer)
//   3: bf16 [B,H,N,64] per-head transposed        (Q, K)
// ---------------------------------------------------------------------------
template <int SMODE>
__global__ __launch_bounds__(128)
void gemm_bf16_kernel(const bf16_t* __restrict__ Wb,
                      const bf16_t* __restrict__ Xb,
                      const float* __restrict__ bias,
                      int Cin, int xld, int Npos,
                      float* __restrict__ outF, bf16_t* __restrict__ outB,
                      const float* __restrict__ resid, int ncStride) {
  const int lane = threadIdx.x & 31;
  const int wave = threadIdx.x >> 5;   // 0..3
  const int l16  = lane & 15;
  const int hh   = lane >> 4;          // 0/1
  const int b    = blockIdx.z;
  const int coutBase = blockIdx.y * 64;
  const int CoutTot  = gridDim.y * 64;
  const int n0 = blockIdx.x * 64 + wave * 16;

  v8f acc[4] = {};

  const bf16_t* xrow  = Xb + ((size_t)b * Npos + (n0 + l16)) * xld;
  const bf16_t* wbase = Wb + (size_t)(coutBase + l16) * Cin;

  for (int k0 = 0; k0 < Cin; k0 += 32) {
    const v16bf bX = frag_2runs(xrow + k0 + 8 * hh, xrow + k0 + 16 + 8 * hh);
#pragma unroll
    for (int mt = 0; mt < 4; ++mt) {
      const bf16_t* wrow = wbase + (size_t)(mt * 16) * Cin;
      const v16bf aW =
          frag_2runs(wrow + k0 + 8 * hh, wrow + k0 + 16 + 8 * hh);
      acc[mt] = wmma_bf16(aW, bX, acc[mt]);
    }
  }

#pragma unroll
  for (int mt = 0; mt < 4; ++mt) {
#pragma unroll
    for (int j = 0; j < 8; ++j) {
      const int o = coutBase + mt * 16 + j + 8 * hh;
      const int n = n0 + l16;
      float v = acc[mt][j] + bias[o];
      if (SMODE == 0) {
        const size_t idx = ((size_t)b * CoutTot + o) * Npos + n;
        outF[idx] = v + resid[idx];
      } else if (SMODE == 1) {
        outB[((size_t)b * CoutTot + o) * Npos + n] = (bf16_t)v;
      } else if (SMODE == 2) {
        outB[((size_t)b * Npos + n) * (size_t)ncStride + o] = (bf16_t)v;
      } else {
        outB[(((size_t)b * gridDim.y + blockIdx.y) * Npos + n) * 64 +
             (o & 63)] = (bf16_t)v;
      }
    }
  }
}

// ---------------------------------------------------------------------------
// Flash attention (per head, D=64). One wave owns a 16-query tile and streams
// keys 64 at a time:
//   - 8 WMMAs for S (4 tiles x K=64 in 2 steps, aQ frags reused),
//   - online softmax: ONLY the row-max needs a per-iteration cross-lane
//     reduction (4 ds_bpermute steps); the denominator is kept as per-lane
//     partial sums and reduced once after the loop,
//   - P -> A-layout via per-wave LDS (72-element row pitch: 16B-aligned
//     ds_load_b128, bank-stride 36 -> conflict-free),
//   - 8 WMMAs for PV (aP frags reused across the 4 d-tiles).
// Q/K: [B,H,pos,64] bf16 (transposed);  V: [B,C,M] bf16 (channel-major).
// Output: attn [B,N,C] bf16 (position-major, ready for the wm GEMM).
// ---------------------------------------------------------------------------
__global__ __launch_bounds__(128)
void flash_attn_kernel(const bf16_t* __restrict__ Qt,
                       const bf16_t* __restrict__ Kt,
                       const bf16_t* __restrict__ Vb,
                       bf16_t* __restrict__ attnOut) {
  const int lane = threadIdx.x & 31;
  const int wave = threadIdx.x >> 5;
  const int l16  = lane & 15;
  const int hh   = lane >> 4;
  const int h    = blockIdx.y;
  const int b    = blockIdx.z;
  const int n0   = (blockIdx.x * 4 + wave) * 16;

  __shared__ __align__(16) bf16_t Pb[4][16][72];

  const bf16_t* qrow = Qt + (((size_t)b * NHEAD + h) * NPOS + n0 + l16) * 64;
  const v16bf aQ0 = frag_2runs(qrow + 8 * hh, qrow + 16 + 8 * hh);
  const v16bf aQ1 = frag_2runs(qrow + 32 + 8 * hh, qrow + 48 + 8 * hh);

  const bf16_t* kbase = Kt + ((size_t)b * NHEAD + h) * (size_t)NPOS * 64;
  const bf16_t* vbase = Vb + ((size_t)b * CCH + h * 64) * (size_t)NPOS;

  float mrow[8], lpart[8];
#pragma unroll
  for (int j = 0; j < 8; ++j) { mrow[j] = -1e30f; lpart[j] = 0.f; }
  v8f acc[4] = {};

  const float qk_scale = 0.125f;  // 1/sqrt(64)

  for (int m0 = 0; m0 < NPOS; m0 += 64) {
    // ---- S = (Q K^T): 4 column tiles of 16, K-dim 64 in two steps ----
    v8f S[4];
#pragma unroll
    for (int ct = 0; ct < 4; ++ct) {
      const bf16_t* krow = kbase + (size_t)(m0 + ct * 16 + l16) * 64;
      v8f s = {};
      s = wmma_bf16(aQ0, frag_2runs(krow + 8 * hh, krow + 16 + 8 * hh), s);
      s = wmma_bf16(aQ1, frag_2runs(krow + 32 + 8 * hh, krow + 48 + 8 * hh), s);
      S[ct] = s;
    }

    // ---- online softmax: row max (only per-iteration reduction) ----
#pragma unroll
    for (int j = 0; j < 8; ++j) {
      float t = fmaxf(fmaxf(S[0][j], S[1][j]), fmaxf(S[2][j], S[3][j]));
      t = fmaxf(t, __shfl_xor(t, 1, 32));
      t = fmaxf(t, __shfl_xor(t, 2, 32));
      t = fmaxf(t, __shfl_xor(t, 4, 32));
      t = fmaxf(t, __shfl_xor(t, 8, 32));
      const float mnew = fmaxf(mrow[j], t * qk_scale);
      const float corr = __expf(mrow[j] - mnew);
      mrow[j] = mnew;
      const float p0 = __expf(S[0][j] * qk_scale - mnew);
      const float p1 = __expf(S[1][j] * qk_scale - mnew);
      const float p2 = __expf(S[2][j] * qk_scale - mnew);
      const float p3 = __expf(S[3][j] * qk_scale - mnew);
      // denominator: per-lane partial only (corr is row-uniform across lanes)
      lpart[j] = lpart[j] * corr + ((p0 + p1) + (p2 + p3));
#pragma unroll
      for (int t4 = 0; t4 < 4; ++t4) acc[t4][j] *= corr;
      Pb[wave][j + 8 * hh][l16]      = (bf16_t)p0;
      Pb[wave][j + 8 * hh][16 + l16] = (bf16_t)p1;
      Pb[wave][j + 8 * hh][32 + l16] = (bf16_t)p2;
      Pb[wave][j + 8 * hh][48 + l16] = (bf16_t)p3;
    }
    // LDS ops from one wave complete in order; just stop the compiler from
    // reordering the ds loads ahead of the stores.
    __builtin_amdgcn_wave_barrier();
    const bf16_t* prow = &Pb[wave][l16][0];
    const v16bf aP0 = frag_2runs(prow + 8 * hh, prow + 16 + 8 * hh);
    const v16bf aP1 = frag_2runs(prow + 32 + 8 * hh, prow + 48 + 8 * hh);
#pragma unroll
    for (int t4 = 0; t4 < 4; ++t4) {
      const bf16_t* vrow = vbase + (size_t)(t4 * 16 + l16) * NPOS + m0;
      acc[t4] = wmma_bf16(aP0, frag_2runs(vrow + 8 * hh, vrow + 16 + 8 * hh), acc[t4]);
      acc[t4] = wmma_bf16(aP1, frag_2runs(vrow + 32 + 8 * hh, vrow + 48 + 8 * hh), acc[t4]);
    }
    __builtin_amdgcn_wave_barrier();
  }

  // ---- single deferred denominator reduction + normalize + store ----
  float inv[8];
#pragma unroll
  for (int j = 0; j < 8; ++j) {
    float rs = lpart[j];
    rs += __shfl_xor(rs, 1, 32);
    rs += __shfl_xor(rs, 2, 32);
    rs += __shfl_xor(rs, 4, 32);
    rs += __shfl_xor(rs, 8, 32);
    inv[j] = 1.f / rs;
  }
#pragma unroll
  for (int t4 = 0; t4 < 4; ++t4) {
#pragma unroll
    for (int j = 0; j < 8; ++j) {
      attnOut[((size_t)b * NPOS + n0 + j + 8 * hh) * CCH + h * 64 + t4 * 16 +
              l16] = (bf16_t)(acc[t4][j] * inv[j]);
    }
  }
}

// ---------------------------------------------------------------------------
// BatchNorm batch statistics: one block per channel (deterministic, no
// atomics). h is bf16 [B, 2C, N].
// ---------------------------------------------------------------------------
__global__ __launch_bounds__(256)
void bn_stats_kernel(const bf16_t* __restrict__ h, float* __restrict__ gsum,
                     float* __restrict__ gsq) {
  const int c = blockIdx.x;  // 0..511
  float s = 0.f, q = 0.f;
  for (int r = threadIdx.x; r < BATCH * NPOS; r += 256) {
    const int b = r >> 12;
    const int n = r & (NPOS - 1);
    const float v = (float)h[((size_t)b * 512 + c) * NPOS + n];
    s += v;
    q += v * v;
  }
  __shared__ float sh[256], shq[256];
  sh[threadIdx.x] = s;
  shq[threadIdx.x] = q;
  __syncthreads();
  for (int w = 128; w > 0; w >>= 1) {
    if (threadIdx.x < w) {
      sh[threadIdx.x] += sh[threadIdx.x + w];
      shq[threadIdx.x] += shq[threadIdx.x + w];
    }
    __syncthreads();
  }
  if (threadIdx.x == 0) {
    gsum[c] = sh[0];
    gsq[c] = shq[0];
  }
}

__global__ void bn_finalize_kernel(const float* __restrict__ gsum,
                                   const float* __restrict__ gsq,
                                   const float* __restrict__ gamma,
                                   const float* __restrict__ beta,
                                   float* __restrict__ scaleArr,
                                   float* __restrict__ shiftArr) {
  const int c = blockIdx.x * blockDim.x + threadIdx.x;
  if (c < 512) {
    const float invCnt = 1.f / (float)(BATCH * NPOS);
    const float mean = gsum[c] * invCnt;
    const float var  = gsq[c] * invCnt - mean * mean;
    const float sc   = gamma[c] * rsqrtf(var + 1e-5f);
    scaleArr[c] = sc;
    shiftArr[c] = beta[c] - mean * sc;
  }
}

// ---------------------------------------------------------------------------
// Host side
// ---------------------------------------------------------------------------
extern "C" void kernel_launch(void* const* d_in, const int* in_sizes, int n_in,
                              void* d_out, int out_size, void* d_ws,
                              size_t ws_size, hipStream_t stream) {
  (void)in_sizes; (void)n_in; (void)out_size; (void)ws_size;
  const float* m1 = (const float*)d_in[0];
  const float* m2 = (const float*)d_in[1];
  const float* wq = (const float*)d_in[2];
  const float* bq = (const float*)d_in[3];
  const float* wk = (const float*)d_in[4];
  const float* bk = (const float*)d_in[5];
  const float* wv = (const float*)d_in[6];
  const float* bv = (const float*)d_in[7];
  const float* wm = (const float*)d_in[8];
  const float* bm = (const float*)d_in[9];
  const float* w1 = (const float*)d_in[10];
  const float* b1 = (const float*)d_in[11];
  const float* gamma = (const float*)d_in[12];
  const float* beta  = (const float*)d_in[13];
  const float* w2 = (const float*)d_in[14];
  const float* b2 = (const float*)d_in[15];
  float* outp = (float*)d_out;

  char* ws = (char*)d_ws;
  size_t off = 0;
  auto take = [&](size_t bytes) -> char* {
    char* p = ws + off;
    off += (bytes + 255) & ~(size_t)255;
    return p;
  };
  const size_t posC  = (size_t)BATCH * NPOS * CCH * sizeof(bf16_t);   // 16.8MB
  bf16_t* Qt    = (bf16_t*)take(posC);        // [B,H,N,64]
  bf16_t* Kt    = (bf16_t*)take(posC);        // [B,H,M,64] (adjacent to Qt)
  bf16_t* Vb    = (bf16_t*)take(posC);        // [B,C,M]
  bf16_t* attnb = (bf16_t*)take(posC);        // [B,N,C]    (adjacent to Vb)
  bf16_t* xcat  = (bf16_t*)take(2 * posC);    // [B,N,512]  (m1 | wm*attn)
  bf16_t* m2bf  = (bf16_t*)take(posC);        // [B,N,C]
  bf16_t* wqbf  = (bf16_t*)take(CCH * CCH * sizeof(bf16_t));
  bf16_t* wkbf  = (bf16_t*)take(CCH * CCH * sizeof(bf16_t));
  bf16_t* wvbf  = (bf16_t*)take(CCH * CCH * sizeof(bf16_t));
  bf16_t* wmbf  = (bf16_t*)take(CCH * CCH * sizeof(bf16_t));
  bf16_t* w1bf  = (bf16_t*)take(512 * 512 * sizeof(bf16_t));
  bf16_t* w2bf  = (bf16_t*)take(CCH * 512 * sizeof(bf16_t));
  float* gsum     = (float*)take(512 * sizeof(float));
  float* gsq      = (float*)take(512 * sizeof(float));
  float* scaleArr = (float*)take(512 * sizeof(float));
  float* shiftArr = (float*)take(512 * sizeof(float));
  // Dead-buffer reuse: h (B,512,N bf16 = 33.5MB) over Qt+Kt; hn over Vb+attnb.
  bf16_t* hbuf = Qt;
  bf16_t* hn   = Vb;

  // 1) weights -> bf16
  cast_f32_bf16_kernel<<<64, 256, 0, stream>>>(wq, wqbf, CCH * CCH);
  cast_f32_bf16_kernel<<<64, 256, 0, stream>>>(wk, wkbf, CCH * CCH);
  cast_f32_bf16_kernel<<<64, 256, 0, stream>>>(wv, wvbf, CCH * CCH);
  cast_f32_bf16_kernel<<<64, 256, 0, stream>>>(wm, wmbf, CCH * CCH);
  cast_f32_bf16_kernel<<<128, 256, 0, stream>>>(w1, w1bf, 512 * 512);
  cast_f32_bf16_kernel<<<128, 256, 0, stream>>>(w2, w2bf, CCH * 512);

  // 2) activations -> position-major bf16 (m1 into channels 0..255 of xcat)
  transpose_cn_nc_kernel<float, false>
      <<<dim3(NPOS / 32, CCH / 32, BATCH), 256, 0, stream>>>(
          m1, xcat, nullptr, nullptr, CCH, NPOS, 512);
  transpose_cn_nc_kernel<float, false>
      <<<dim3(NPOS / 32, CCH / 32, BATCH), 256, 0, stream>>>(
          m2, m2bf, nullptr, nullptr, CCH, NPOS, CCH);

  // 3) Q,K (head-transposed), V (channel-major)
  dim3 gemmGrid(NPOS / 64, CCH / 64, BATCH);
  gemm_bf16_kernel<3><<<gemmGrid, 128, 0, stream>>>(
      wqbf, xcat, bq, CCH, 512, NPOS, nullptr, Qt, nullptr, 0);
  gemm_bf16_kernel<3><<<gemmGrid, 128, 0, stream>>>(
      wkbf, m2bf, bk, CCH, CCH, NPOS, nullptr, Kt, nullptr, 0);
  gemm_bf16_kernel<1><<<gemmGrid, 128, 0, stream>>>(
      wvbf, m2bf, bv, CCH, CCH, NPOS, nullptr, Vb, nullptr, 0);

  // 4) flash attention -> attnb [B,N,C]
  flash_attn_kernel<<<dim3(NPOS / 64, NHEAD, BATCH), 128, 0, stream>>>(
      Qt, Kt, Vb, attnb);

  // 5) wm projection -> channels 256..511 of xcat (free concat)
  gemm_bf16_kernel<2><<<gemmGrid, 128, 0, stream>>>(
      wmbf, attnb, bm, CCH, CCH, NPOS, nullptr, xcat + 256, nullptr, 512);

  // 6) w1: [512x512] x [B,N,512] -> h [B,512,N] bf16
  gemm_bf16_kernel<1><<<dim3(NPOS / 64, 512 / 64, BATCH), 128, 0, stream>>>(
      w1bf, xcat, b1, 512, 512, NPOS, nullptr, hbuf, nullptr, 0);

  // 7) BatchNorm stats (deterministic) + finalize
  bn_stats_kernel<<<512, 256, 0, stream>>>(hbuf, gsum, gsq);
  bn_finalize_kernel<<<2, 256, 0, stream>>>(gsum, gsq, gamma, beta, scaleArr,
                                            shiftArr);

  // 8) fused BN+ReLU + transpose -> hn [B,N,512] bf16
  transpose_cn_nc_kernel<bf16_t, true>
      <<<dim3(NPOS / 32, 512 / 32, BATCH), 256, 0, stream>>>(
          hbuf, hn, scaleArr, shiftArr, 512, NPOS, 512);

  // 9) final: w2 x hn + b2 + motion1 -> d_out [B,256,N] f32
  gemm_bf16_kernel<0><<<dim3(NPOS / 64, CCH / 64, BATCH), 128, 0, stream>>>(
      w2bf, hn, b2, 512, 512, NPOS, outp, nullptr, m1, 0);
}
